// AttenNetVLAD_82617990906406
// MI455X (gfx1250) — matrix operand order repaired and checked by reading, not compiled, not run on hardware
//
#include <hip/hip_runtime.h>
#include <hip/hip_bf16.h>
#include <math.h>

// Problem constants (fixed by setup_inputs)
#define NN   48
#define CCH  512
#define PPX  1600
#define KCL  64
#define EPSF 1e-12f

typedef __attribute__((ext_vector_type(16))) __bf16 v16bf;
typedef __attribute__((ext_vector_type(8)))  float  v8f;

union Frag { uint4 q[2]; v16bf v; };

__device__ __forceinline__ unsigned short f2bf(float f) {
  unsigned u = __float_as_uint(f);
  u += 0x7FFFu + ((u >> 16) & 1u);           // round-to-nearest-even
  return (unsigned short)(u >> 16);
}

// ---------------------------------------------------------------- kernel 0
// conv_w [64,512] f32 -> bf16 (stays hot in L2)
__global__ void k_cw(const float* __restrict__ w, unsigned short* __restrict__ cwb) {
  int i = blockIdx.x * 256 + threadIdx.x;
  if (i < KCL * CCH) cwb[i] = f2bf(w[i]);
}

// ---------------------------------------------------------------- kernel 1
// Per pixel: sum of squares (for 1/||x||), attention heatmap dot, x -> bf16.
__global__ void __launch_bounds__(256) k_prep(
    const float* __restrict__ x, const float* __restrict__ aw,
    const float* __restrict__ ab, unsigned short* __restrict__ xb,
    float* __restrict__ rnorm, float* __restrict__ hmp) {
  int t = blockIdx.x * 256 + threadIdx.x;          // 0 .. N*P-1 (exact)
  int n = t / PPX, p = t - n * PPX;
  const float* xr = x + (size_t)n * CCH * PPX + p;
  unsigned short* xw = xb + (size_t)n * CCH * PPX + p;
  float ss = 0.f, at = 0.f;
  #pragma unroll 8
  for (int c = 0; c < CCH; ++c) {
    float v = xr[(size_t)c * PPX];
    ss = fmaf(v, v, ss);
    at = fmaf(fmaxf(v, 0.f), aw[c], at);
    xw[(size_t)c * PPX] = f2bf(v);
  }
  rnorm[t] = 1.f / fmaxf(sqrtf(ss), EPSF);
  hmp[t]   = fmaxf(at + ab[0], 0.f);
}

// ---------------------------------------------------------------- kernel 2
// logits[k,p] = conv_w[k,:] . x_bf16[:,p]; softmax over K; * heatmap;
// emit wr = bf16(w * rnorm), wsum atomics.
// Block: n x (64 pixels). 8 waves: mt = k-tile (0..3), ng = pixel-group (0..1),
// each wave does TWO wmma per step sharing one A fragment.
__global__ void __launch_bounds__(256) k_assign(
    const unsigned short* __restrict__ xb, const unsigned short* __restrict__ cwb,
    const float* __restrict__ rnorm, const float* __restrict__ hmp,
    unsigned short* __restrict__ wr, float* __restrict__ wsum) {
  const int n  = blockIdx.y;
  const int p0 = blockIdx.x * 64;
  const int tid = threadIdx.x;
  const int lane = tid & 31, wid = tid >> 5;
  const int mt = wid & 3, ng = wid >> 2;
  const int nt0 = ng * 2, nt1 = ng * 2 + 1;
  const int hi = lane >> 4, rl = lane & 15;

  // double-buffered transposed tile: [p][c], pitch 40 halfs (80B, 16B-aligned rows)
  __shared__ __align__(16) unsigned short ldsBT[2][64 * 40];
  __shared__ float ldsL[64 * 64];                  // logits / weights

  const unsigned short* xbase = xb + (size_t)n * CCH * PPX + p0;
  const int sc = tid >> 3;                         // staging c-row 0..31
  const int sp = (tid & 7) * 8;                    // staging p-chunk (8 halfs)

  // stage step 0 into buffer 0 (one b128 per thread, transpose on LDS store)
  {
    union { uint4 q; unsigned short h[8]; } u;
    u.q = *(const uint4*)(xbase + (size_t)sc * PPX + sp);
    #pragma unroll
    for (int j = 0; j < 8; ++j) ldsBT[0][(sp + j) * 40 + sc] = u.h[j];
  }
  __syncthreads();

  v8f acc0 = {}, acc1 = {};
  for (int step = 0; step < 16; ++step) {
    const unsigned short* cur = ldsBT[step & 1];
    if (step < 15) {                               // stage next tile (other buffer)
      union { uint4 q; unsigned short h[8]; } u;
      u.q = *(const uint4*)(xbase + (size_t)((step + 1) * 32 + sc) * PPX + sp);
      unsigned short* nb = ldsBT[(step + 1) & 1];
      #pragma unroll
      for (int j = 0; j < 8; ++j) nb[(sp + j) * 40 + sc] = u.h[j];
    }
    Frag a, b0, b1;
    // A frag (16-bit A layout): row k, c-runs {0..7,16..23} + (lane>=16 ? 8)
    const uint4* ap = (const uint4*)(cwb + (size_t)(mt * 16 + rl) * CCH + step * 32 + hi * 8);
    a.q[0] = ap[0];
    a.q[1] = ap[2];
    // B frags: n-col = pixel, c-runs {0..7,16..23} + (lane>=16 ? 8)
    const uint4* bp0 = (const uint4*)(cur + (nt0 * 16 + rl) * 40 + hi * 8);
    b0.q[0] = bp0[0];
    b0.q[1] = bp0[2];
    const uint4* bp1 = (const uint4*)(cur + (nt1 * 16 + rl) * 40 + hi * 8);
    b1.q[0] = bp1[0];
    b1.q[1] = bp1[2];
    acc0 = __builtin_amdgcn_wmma_f32_16x16x32_bf16(false, a.v, false, b0.v,
                                                   (short)0, acc0, false, false);
    acc1 = __builtin_amdgcn_wmma_f32_16x16x32_bf16(false, a.v, false, b1.v,
                                                   (short)0, acc1, false, false);
    __syncthreads();
  }
  // C/D layout: element j -> (M = mt*16 + hi*8 + j, N = lane&15)
  #pragma unroll
  for (int j = 0; j < 8; ++j) {
    ldsL[(mt * 16 + hi * 8 + j) * 64 + nt0 * 16 + rl] = acc0[j];
    ldsL[(mt * 16 + hi * 8 + j) * 64 + nt1 * 16 + rl] = acc1[j];
  }
  __syncthreads();

  if (tid < 64) {                      // one thread per pixel: softmax over K=64
    const int p = tid, gp = p0 + p;
    const float rn = rnorm[n * PPX + gp];
    const float hm = hmp[n * PPX + gp];
    float m = -3.4e38f;
    for (int k = 0; k < KCL; ++k) m = fmaxf(m, ldsL[k * 64 + p] * rn);
    float s = 0.f;
    for (int k = 0; k < KCL; ++k) {
      float e = __expf(ldsL[k * 64 + p] * rn - m);
      ldsL[k * 64 + p] = e; s += e;
    }
    const float scale = hm / s;
    for (int k = 0; k < KCL; ++k) {
      float w = ldsL[k * 64 + p] * scale;
      ldsL[k * 64 + p] = w;                                   // for wsum
      wr[((size_t)n * KCL + k) * PPX + gp] = f2bf(w * rn);    // fused rnorm
    }
  }
  __syncthreads();
  if (tid < KCL) {                     // per-k partial of wsum, 1 atomic each
    float s = 0.f;
    #pragma unroll
    for (int p = 0; p < 64; ++p) s += ldsL[tid * 64 + p];
    atomicAdd(&wsum[n * KCL + tid], s);
  }
}

// ---------------------------------------------------------------- kernel 3
// term1[k,c] = sum_p wr[k,p] * xb[c,p]   (reduction over P=1600)
// Both fragments load straight from global: both operands are p-contiguous.
__global__ void __launch_bounds__(256) k_aggregate(
    const unsigned short* __restrict__ xb, const unsigned short* __restrict__ wr,
    float* __restrict__ term1) {
  const int n = blockIdx.y;
  const int ct = blockIdx.x;                       // c-tile of 32
  const int tid = threadIdx.x;
  const int lane = tid & 31, wid = tid >> 5;
  const int mt = wid & 3, nt = wid >> 2;
  const int hi = lane >> 4, rl = lane & 15;

  const int k = mt * 16 + rl;                      // A row
  const int c = ct * 32 + nt * 16 + rl;            // B column
  const unsigned short* arow = wr + ((size_t)n * KCL + k) * PPX + hi * 8;
  const unsigned short* brow = xb + ((size_t)n * CCH + c) * PPX + hi * 8;

  v8f acc = {};
  #pragma unroll 2
  for (int pp = 0; pp < PPX; pp += 32) {
    __builtin_prefetch(arow + pp + 512, 0, 0);     // global_prefetch ~1KB ahead
    __builtin_prefetch(brow + pp + 512, 0, 0);
    Frag a, b;
    const uint4* ap = (const uint4*)(arow + pp);
    a.q[0] = ap[0];
    a.q[1] = ap[2];
    const uint4* bp = (const uint4*)(brow + pp);
    b.q[0] = bp[0];
    b.q[1] = bp[2];
    acc = __builtin_amdgcn_wmma_f32_16x16x32_bf16(false, a.v, false, b.v,
                                                  (short)0, acc, false, false);
  }
  #pragma unroll
  for (int j = 0; j < 8; ++j) {
    int kk = mt * 16 + hi * 8 + j;
    term1[((size_t)n * KCL + kk) * CCH + c] = acc[j];
  }
}

// ---------------------------------------------------------------- kernel 4a
// vlad[n,k,:] = term1 - wsum*centroid ; intra L2-normalize ; accumulate global ss
__global__ void __launch_bounds__(256) k_intra(
    const float* __restrict__ term1, const float* __restrict__ wsum,
    const float* __restrict__ cent, float* __restrict__ out,
    float* __restrict__ gss) {
  const int n = blockIdx.y, k = blockIdx.x;
  const int t = threadIdx.x;
  const float ws = wsum[n * KCL + k];
  const float* tr = term1 + ((size_t)n * KCL + k) * CCH;
  const float* cr = cent + (size_t)k * CCH;
  float v0 = tr[t]       - ws * cr[t];
  float v1 = tr[t + 256] - ws * cr[t + 256];
  float ss = v0 * v0 + v1 * v1;
  __shared__ float red[256];
  __shared__ float rsh;
  red[t] = ss; __syncthreads();
  for (int s = 128; s > 0; s >>= 1) {
    if (t < s) red[t] += red[t + s];
    __syncthreads();
  }
  if (t == 0) {
    float tot = red[0];
    float rn = 1.f / fmaxf(sqrtf(tot), EPSF);
    rsh = rn;
    atomicAdd(&gss[n], tot * rn * rn);
  }
  __syncthreads();
  float rn = rsh;
  float* o = out + ((size_t)n * KCL + k) * CCH;
  o[t]       = v0 * rn;
  o[t + 256] = v1 * rn;
}

// ---------------------------------------------------------------- kernel 4b
__global__ void __launch_bounds__(256) k_global(float* __restrict__ out,
                                                const float* __restrict__ gss) {
  const int n = blockIdx.y;
  const int i = blockIdx.x * 256 + threadIdx.x;    // 0..32767 exact
  const float g = 1.f / fmaxf(sqrtf(gss[n]), EPSF);
  out[(size_t)n * (KCL * CCH) + i] *= g;
}

// ---------------------------------------------------------------- launch
// Workspace layout (bytes, 256-aligned)
#define OFF_CWB   ((size_t)0)
#define OFF_XB    ((size_t)65536)
#define OFF_WR    (OFF_XB + (size_t)NN * CCH * PPX * 2)     // +78,643,200
#define OFF_RN    (OFF_WR + (size_t)NN * KCL * PPX * 2)     // + 9,830,400
#define OFF_HM    (OFF_RN + (size_t)NN * PPX * 4)           // +   307,200
#define OFF_T1    (OFF_HM + (size_t)NN * PPX * 4)           // +   307,200
#define OFF_WSUM  (OFF_T1 + (size_t)NN * KCL * CCH * 4)     // + 6,291,456
#define OFF_GSS   (OFF_WSUM + (size_t)NN * KCL * 4)         // +    12,288
// total ~95.5 MB

extern "C" void kernel_launch(void* const* d_in, const int* in_sizes, int n_in,
                              void* d_out, int out_size, void* d_ws, size_t ws_size,
                              hipStream_t stream) {
  (void)in_sizes; (void)n_in; (void)out_size; (void)ws_size;
  const float* x   = (const float*)d_in[0];
  const float* cw  = (const float*)d_in[1];
  const float* aw  = (const float*)d_in[2];
  const float* ab  = (const float*)d_in[3];
  const float* cen = (const float*)d_in[4];
  float* out = (float*)d_out;

  char* ws = (char*)d_ws;
  unsigned short* cwb = (unsigned short*)(ws + OFF_CWB);
  unsigned short* xb  = (unsigned short*)(ws + OFF_XB);
  unsigned short* wrb = (unsigned short*)(ws + OFF_WR);
  float* rnorm = (float*)(ws + OFF_RN);
  float* hmp   = (float*)(ws + OFF_HM);
  float* term1 = (float*)(ws + OFF_T1);
  float* wsum  = (float*)(ws + OFF_WSUM);
  float* gss   = (float*)(ws + OFF_GSS);

  // zero accumulators (wsum + gss are adjacent)
  hipMemsetAsync(ws + OFF_WSUM, 0, (size_t)NN * KCL * 4 + 256, stream);

  k_cw<<<dim3((KCL * CCH + 255) / 256), 256, 0, stream>>>(cw, cwb);
  k_prep<<<dim3((NN * PPX) / 256), 256, 0, stream>>>(x, aw, ab, xb, rnorm, hmp);
  k_assign<<<dim3(PPX / 64, NN), 256, 0, stream>>>(xb, cwb, rnorm, hmp, wrb, wsum);
  k_aggregate<<<dim3(CCH / 32, NN), 256, 0, stream>>>(xb, wrb, term1);
  k_intra<<<dim3(KCL, NN), 256, 0, stream>>>(term1, wsum, cen, out, gss);
  k_global<<<dim3(KCL * CCH / 256, NN), 256, 0, stream>>>(out, gss);
}
